// MPSelfAttentionTorch_79130477462196
// MI455X (gfx1250) — matrix operand order repaired and checked by reading, not compiled, not run on hardware
//
#include <hip/hip_runtime.h>

#define DIMN 1536
#define NHEADS 12
#define HD 128
#define SEQ 2304
#define EPSV 1e-6f

#define AS1 __attribute__((address_space(1)))
#define AS3 __attribute__((address_space(3)))

#if __has_builtin(__builtin_amdgcn_global_load_async_to_lds_b128) && \
    __has_builtin(__builtin_amdgcn_s_wait_asynccnt)
#define HAVE_ASYNC_LDS 1
#else
#define HAVE_ASYNC_LDS 0
#endif

typedef _Float16 half_t;
typedef __attribute__((ext_vector_type(16))) _Float16 v16h;
typedef __attribute__((ext_vector_type(8)))  _Float16 v8h;
typedef __attribute__((ext_vector_type(8)))  float    v8f;
typedef int v4i_vs __attribute__((vector_size(4 * sizeof(int))));

static __device__ __forceinline__ v16h make_v16(v8h lo, v8h hi) {
  union { v16h v; v8h h[2]; } u;
  u.h[0] = lo; u.h[1] = hi;
  return u.v;
}

// A fragment 16x32 (MxK), wave32 layout (ISA 7.12.2):
// lane l: row m=l%16; halves0..7 = A[m][k0+8*(l/16)..+8); halves8..15 = A[m][k0+16+8*(l/16)..+8)
static __device__ __forceinline__ v16h load_frag_a(const half_t* p, int row0, int k0, int ld) {
  int lane = threadIdx.x & 31;
  int m = lane & 15, g = lane >> 4;
  const half_t* r = p + (size_t)(row0 + m) * ld + k0 + 8 * g;
  v8h lo = *(const v8h*)(r);
  v8h hi = *(const v8h*)(r + 16);
  return make_v16(lo, hi);
}

// B fragment 32x16 (KxN) where memory holds Bt row-major: Bt[n][k] = B[k][n].
// lane l: col n=l%16; halves0..7 = Bt[n][k0+8*(l/16)..+8); halves8..15 = Bt[n][k0+16+8*(l/16)..+8)
static __device__ __forceinline__ v16h load_frag_b(const half_t* p, int n0, int k0, int ld) {
  int lane = threadIdx.x & 31;
  int n = lane & 15, g = lane >> 4;
  const half_t* r = p + (size_t)(n0 + n) * ld + k0 + 8 * g;
  v8h lo = *(const v8h*)(r);
  v8h hi = *(const v8h*)(r + 16);
  return make_v16(lo, hi);
}

static __device__ __forceinline__ v8f wmma_f16(v16h a, v16h b, v8f c) {
  return __builtin_amdgcn_wmma_f32_16x16x32_f16(false, a, false, b, (short)0, c, false, false);
}

// ---------------- elementwise converts ----------------
__global__ void cvt_f32_f16(const float* __restrict__ in, half_t* __restrict__ out, int n) {
  int i = blockIdx.x * blockDim.x + threadIdx.x;
  if (i < n) out[i] = (half_t)in[i];
}

__global__ void v_transpose(const float* __restrict__ v, half_t* __restrict__ vt) {
  int i = blockIdx.x * blockDim.x + threadIdx.x;
  if (i < SEQ * DIMN) {
    int s = i / DIMN, col = i % DIMN;
    vt[(size_t)col * SEQ + s] = (half_t)v[i];   // [head*128+d][s]
  }
}

// ---------------- RoPE angle tables ----------------
__global__ void rope_table(const float* __restrict__ theta, float* __restrict__ ctab,
                           float* __restrict__ stab) {
  int t = blockIdx.x * blockDim.x + threadIdx.x;
  if (t >= SEQ * 64) return;
  int s = t >> 6, cc = t & 63;
  int wi = s % 24, hh = (s / 24) % 16, ff = s / (24 * 16);
  int pos = (cc < 22) ? ff : ((cc < 43) ? hh : wi);   // t0=22, t1=21
  float a = theta[pos * 64 + cc];
  ctab[t] = cosf(a);
  stab[t] = sinf(a);
}

// ---------------- GEMM: C[S x DIMN] = A[S x DIMN](f16) @ W[DIMN x DIMN]^T(f16) + bias ----------
// wave tile: 32 rows x 64 cols (2 M-subtiles x 4 N-subtiles; B-frags reused across M)
__global__ __launch_bounds__(256) void gemm_xwT(const half_t* __restrict__ A,
                                                const half_t* __restrict__ W,
                                                const float* __restrict__ bias,
                                                float* __restrict__ C) {
  int wave = threadIdx.x >> 5;
  int lane = threadIdx.x & 31;
  int n = lane & 15, g = lane >> 4;
  int row0 = blockIdx.x * 32;
  int col0 = blockIdx.y * 512 + wave * 64;

  v8f acc[2][4];
#pragma unroll
  for (int m = 0; m < 2; m++)
#pragma unroll
    for (int t = 0; t < 4; t++) acc[m][t] = (v8f)(0.f);

  for (int k0 = 0; k0 < DIMN; k0 += 32) {
    if (k0 + 256 < DIMN) {  // stream-ahead hint -> global_prefetch_b8
      __builtin_prefetch((const void*)(A + (size_t)(row0 + (lane & 15)) * DIMN + k0 + 256), 0, 1);
    }
    v16h a0 = load_frag_a(A, row0, k0, DIMN);
    v16h a1 = load_frag_a(A, row0 + 16, k0, DIMN);
#pragma unroll
    for (int t = 0; t < 4; t++) {
      v16h b = load_frag_b(W, col0 + 16 * t, k0, DIMN);
      acc[0][t] = wmma_f16(a0, b, acc[0][t]);
      acc[1][t] = wmma_f16(a1, b, acc[1][t]);
    }
  }
#pragma unroll
  for (int m = 0; m < 2; m++) {
#pragma unroll
    for (int t = 0; t < 4; t++) {
      int col = col0 + 16 * t + n;
      float bv = bias[col];
#pragma unroll
      for (int r = 0; r < 8; r++)
        C[(size_t)(row0 + 16 * m + r + 8 * g) * DIMN + col] = acc[m][t][r] + bv;
    }
  }
}

// ---------------- RMSNorm + RoPE -> f16 ----------------
__global__ __launch_bounds__(256) void norm_rope(const float* __restrict__ in,
                                                 const float* __restrict__ gain,
                                                 const float* __restrict__ ctab,
                                                 const float* __restrict__ stab,
                                                 half_t* __restrict__ out,
                                                 float post_scale) {
  int s = blockIdx.x;
  float ss = 0.f;
  for (int i = threadIdx.x; i < DIMN; i += 256) {
    float v = in[(size_t)s * DIMN + i];
    ss += v * v;
  }
#pragma unroll
  for (int m = 16; m >= 1; m >>= 1) ss += __shfl_xor(ss, m, 32);
  __shared__ float wsum[8];
  if ((threadIdx.x & 31) == 0) wsum[threadIdx.x >> 5] = ss;
  __syncthreads();
  float tot = 0.f;
#pragma unroll
  for (int w = 0; w < 8; w++) tot += wsum[w];
  float rms = rsqrtf(tot / (float)DIMN + EPSV);

  for (int p = threadIdx.x; p < DIMN / 2; p += 256) {
    int h = p >> 6, cc = p & 63;
    int base = h * 128 + 2 * cc;
    float xr = in[(size_t)s * DIMN + base] * rms * gain[base];
    float xi = in[(size_t)s * DIMN + base + 1] * rms * gain[base + 1];
    float c = ctab[s * 64 + cc], si = stab[s * 64 + cc];
    out[(size_t)s * DIMN + base]     = (half_t)((xr * c - xi * si) * post_scale);
    out[(size_t)s * DIMN + base + 1] = (half_t)((xr * si + xi * c) * post_scale);
  }
}

// ---------------- flash attention ----------------
// block = 4 waves (128 thr), one head, 64 query rows (16 per wave).
// Per 32-key iteration: K-tile (32x128) and V-tile (128x32) staged into LDS via
// GLOBAL_LOAD_ASYNC_TO_LDS_B128 (ASYNCcnt) when available; fragments then built
// from LDS (4x less global traffic than per-wave fragment loads).
__global__ __launch_bounds__(128) void attn_flash(const half_t* __restrict__ qh,
                                                  const half_t* __restrict__ kh,
                                                  const half_t* __restrict__ vt,
                                                  half_t* __restrict__ ah) {
  __shared__ __align__(16) half_t Kt[32 * 128];      // [key][dim]
  __shared__ __align__(16) half_t Vs[128 * 32];      // [dim][key]
  __shared__ __align__(16) half_t Pbuf[4][16 * 32];  // per-wave P
  int wave = threadIdx.x >> 5;
  int lane = threadIdx.x & 31;
  int n = lane & 15, g = lane >> 4;
  int head = blockIdx.y;
  int qrow = blockIdx.x * 64 + wave * 16;
  const half_t* vhead = vt + (size_t)head * HD * SEQ;
  half_t* P = &Pbuf[wave][0];

  v16h qa[4];
#pragma unroll
  for (int t = 0; t < 4; t++) qa[t] = load_frag_a(qh, qrow, head * HD + 32 * t, DIMN);

  v8f acc[8];
#pragma unroll
  for (int t = 0; t < 8; t++) acc[t] = (v8f)(0.f);
  float mi[8], li[8];
#pragma unroll
  for (int r = 0; r < 8; r++) { mi[r] = -3.0e38f; li[r] = 0.f; }

  int kr = threadIdx.x >> 2, seg = threadIdx.x & 3;  // K-tile: row 0..31, 32-half segment
  int vd = threadIdx.x;                              // V-tile: dim 0..127

  for (int k0 = 0; k0 < SEQ; k0 += 32) {
    __syncthreads();  // previous iteration's LDS reads done
    {
      const half_t* srcK = kh + (size_t)(k0 + kr) * DIMN + head * HD + seg * 32;
      half_t* dstK = Kt + kr * 128 + seg * 32;
      const half_t* srcV = vhead + (size_t)vd * SEQ + k0;
      half_t* dstV = Vs + vd * 32;
#if HAVE_ASYNC_LDS
#pragma unroll
      for (int q = 0; q < 4; q++) {
        __builtin_amdgcn_global_load_async_to_lds_b128(
            (AS1 v4i_vs*)(srcK + 8 * q), (AS3 v4i_vs*)(dstK + 8 * q), 0, 0);
        __builtin_amdgcn_global_load_async_to_lds_b128(
            (AS1 v4i_vs*)(srcV + 8 * q), (AS3 v4i_vs*)(dstV + 8 * q), 0, 0);
      }
      __builtin_amdgcn_s_wait_asynccnt(0);
#else
      const v8h* sk = (const v8h*)srcK; v8h* dk = (v8h*)dstK;
      dk[0] = sk[0]; dk[1] = sk[1]; dk[2] = sk[2]; dk[3] = sk[3];
      const v8h* sv = (const v8h*)srcV; v8h* dv = (v8h*)dstV;
      dv[0] = sv[0]; dv[1] = sv[1]; dv[2] = sv[2]; dv[3] = sv[3];
#endif
    }
    __syncthreads();

    v8f s0 = (v8f)(0.f), s1 = (v8f)(0.f);
#pragma unroll
    for (int t = 0; t < 4; t++) {
      v16h b0 = load_frag_b(Kt, 0, 32 * t, HD);
      s0 = wmma_f16(qa[t], b0, s0);
      v16h b1 = load_frag_b(Kt, 16, 32 * t, HD);
      s1 = wmma_f16(qa[t], b1, s1);
    }
    float corr[8];
#pragma unroll
    for (int r = 0; r < 8; r++) {
      float mx = fmaxf(s0[r], s1[r]);
#pragma unroll
      for (int m = 8; m >= 1; m >>= 1) mx = fmaxf(mx, __shfl_xor(mx, m, 32));
      float mn = fmaxf(mi[r], mx);
      corr[r] = __expf(mi[r] - mn);
      mi[r] = mn;
    }
#pragma unroll
    for (int r = 0; r < 8; r++) {
      float p0 = __expf(s0[r] - mi[r]);
      float p1 = __expf(s1[r] - mi[r]);
      float rs = p0 + p1;
#pragma unroll
      for (int m = 8; m >= 1; m >>= 1) rs += __shfl_xor(rs, m, 32);
      li[r] = li[r] * corr[r] + rs;
      P[(r + 8 * g) * 32 + n]      = (half_t)p0;
      P[(r + 8 * g) * 32 + 16 + n] = (half_t)p1;
    }
#pragma unroll
    for (int t = 0; t < 8; t++)
#pragma unroll
      for (int r = 0; r < 8; r++) acc[t][r] *= corr[r];

    // same-wave LDS ops are in-order; reload P in A-fragment layout
    v16h pa = load_frag_a(P, 0, 0, 32);
#pragma unroll
    for (int t = 0; t < 8; t++) {
      v16h vb = load_frag_b(Vs, t * 16, 0, 32);
      acc[t] = wmma_f16(pa, vb, acc[t]);
    }
  }
#pragma unroll
  for (int t = 0; t < 8; t++) {
#pragma unroll
    for (int r = 0; r < 8; r++) {
      float o = acc[t][r] / li[r];
      ah[(size_t)(qrow + r + 8 * g) * DIMN + head * HD + t * 16 + n] = (half_t)o;
    }
  }
}

// ---------------- host ----------------
extern "C" void kernel_launch(void* const* d_in, const int* in_sizes, int n_in,
                              void* d_out, int out_size, void* d_ws, size_t ws_size,
                              hipStream_t stream) {
  (void)in_sizes; (void)n_in; (void)out_size; (void)ws_size;
  const float* x     = (const float*)d_in[0];
  const float* theta = (const float*)d_in[1];
  const float* wq = (const float*)d_in[2];  const float* bq = (const float*)d_in[3];
  const float* wk = (const float*)d_in[4];  const float* bk = (const float*)d_in[5];
  const float* wv = (const float*)d_in[6];  const float* bv = (const float*)d_in[7];
  const float* wo = (const float*)d_in[8];  const float* bo = (const float*)d_in[9];
  const float* gq = (const float*)d_in[10]; const float* gk = (const float*)d_in[11];
  float* out = (float*)d_out;

  const size_t SD  = (size_t)SEQ * DIMN;
  const size_t DD  = (size_t)DIMN * DIMN;
  char* p = (char*)d_ws;
  half_t* xh  = (half_t*)p; p += SD * 2;
  half_t* wqh = (half_t*)p; p += DD * 2;
  half_t* wkh = (half_t*)p; p += DD * 2;
  half_t* wvh = (half_t*)p; p += DD * 2;
  half_t* woh = (half_t*)p; p += DD * 2;
  float*  qf  = (float*)p;  p += SD * 4;
  float*  kf  = (float*)p;  p += SD * 4;
  float*  vf  = (float*)p;  p += SD * 4;
  half_t* qh  = (half_t*)p; p += SD * 2;
  half_t* kh  = (half_t*)p; p += SD * 2;
  half_t* vt  = (half_t*)p; p += SD * 2;
  half_t* ah  = (half_t*)p; p += SD * 2;
  float*  ctb = (float*)p;  p += (size_t)SEQ * 64 * 4;
  float*  stb = (float*)p;

  const int TPB = 256;
  cvt_f32_f16<<<(int)((SD + TPB - 1) / TPB), TPB, 0, stream>>>(x, xh, (int)SD);
  cvt_f32_f16<<<(int)((DD + TPB - 1) / TPB), TPB, 0, stream>>>(wq, wqh, (int)DD);
  cvt_f32_f16<<<(int)((DD + TPB - 1) / TPB), TPB, 0, stream>>>(wk, wkh, (int)DD);
  cvt_f32_f16<<<(int)((DD + TPB - 1) / TPB), TPB, 0, stream>>>(wv, wvh, (int)DD);
  cvt_f32_f16<<<(int)((DD + TPB - 1) / TPB), TPB, 0, stream>>>(wo, woh, (int)DD);
  rope_table<<<(SEQ * 64 + TPB - 1) / TPB, TPB, 0, stream>>>(theta, ctb, stb);

  dim3 ggrid(SEQ / 32, DIMN / 512);
  gemm_xwT<<<ggrid, 256, 0, stream>>>(xh, wqh, bq, qf);
  gemm_xwT<<<ggrid, 256, 0, stream>>>(xh, wkh, bk, kf);
  gemm_xwT<<<ggrid, 256, 0, stream>>>(xh, wvh, bv, vf);

  const float qscale = 0.08838834764831845f;  // HD^-0.5
  norm_rope<<<SEQ, 256, 0, stream>>>(qf, gq, ctb, stb, qh, qscale);
  norm_rope<<<SEQ, 256, 0, stream>>>(kf, gk, ctb, stb, kh, 1.0f);
  v_transpose<<<(int)((SD + TPB - 1) / TPB), TPB, 0, stream>>>(vf, vt);

  attn_flash<<<dim3(SEQ / 64, NHEADS), 128, 0, stream>>>(qh, kh, vt, ah);

  gemm_xwT<<<ggrid, 256, 0, stream>>>(ah, woh, bo, out);
}
